// WanAttention_67491116089399
// MI455X (gfx1250) — compile-verified, matrix operand order checked by reading, not run on hardware
//
#include <hip/hip_runtime.h>

// ---------------------------------------------------------------------------
// Types for CDNA5 WMMA (wave32): v_wmma_f32_16x16x32_bf16
// ---------------------------------------------------------------------------
typedef __attribute__((ext_vector_type(16))) __bf16 v16bf;
typedef __attribute__((ext_vector_type(8)))  __bf16 v8bf;
typedef __attribute__((ext_vector_type(8)))  float  v8f;
typedef __attribute__((ext_vector_type(4)))  float  v4f;

union BF16x16 { v16bf v; v8bf h[2]; };

__device__ __forceinline__ unsigned short bfbits(float f) {
  unsigned u = __builtin_bit_cast(unsigned, f);
  u += 0x7fffu + ((u >> 16) & 1u);          // round-to-nearest-even
  return (unsigned short)(u >> 16);
}
__device__ __forceinline__ __bf16 f2bf(float f) {
  unsigned short b = bfbits(f);
  return __builtin_bit_cast(__bf16, b);
}

// A/B fragment for 16-bit WMMA; operand rows are K-contiguous in memory.
// lanes 0-15:  K = {0..7, 16..23};  lanes 16-31: K = {8..15, 24..31}
__device__ __forceinline__ v16bf frag_bf(const __bf16* base, int hi) {
  BF16x16 u;
  u.h[0] = *(const v8bf*)(base + hi * 8);
  u.h[1] = *(const v8bf*)(base + 16 + hi * 8);
  return u.v;
}

__device__ __forceinline__ v8f wmma_bf16(v16bf a, v16bf b, v8f c) {
  return __builtin_amdgcn_wmma_f32_16x16x32_bf16(
      /*neg_a=*/false, a, /*neg_b=*/false, b,
      /*c_mod=*/(short)0, c, /*reuse_a=*/false, /*reuse_b=*/false);
}

// ---------------------------------------------------------------------------
// Bulk fp32 -> bf16 conversion (one-time, so GEMM loops are pure bf16 loads).
// n must be a multiple of 2048 (true for all buffers here).
// ---------------------------------------------------------------------------
__global__ __launch_bounds__(256) void cvt_bf16_k(const float* __restrict__ in,
                                                  __bf16* __restrict__ out,
                                                  long n) {
  long i = ((long)blockIdx.x * 256 + threadIdx.x) * 8;
  if (i + 8 <= n) {
    v4f a = *(const v4f*)(in + i);
    v4f b = *(const v4f*)(in + i + 4);
    v8bf r;
    r[0] = f2bf(a[0]); r[1] = f2bf(a[1]); r[2] = f2bf(a[2]); r[3] = f2bf(a[3]);
    r[4] = f2bf(b[0]); r[5] = f2bf(b[1]); r[6] = f2bf(b[2]); r[7] = f2bf(b[3]);
    *(v8bf*)(out + i) = r;
  }
}

// ---------------------------------------------------------------------------
// GEMM: Y[M,N] = X[M,K] @ W[N,K]^T + bias   (X, W already bf16)
// 4 waves per block; each wave computes a 32(M) x 64(N) tile:
// 2 A-fragments x 4 B-fragments -> 8 WMMA per K-step of 32.
// OMODE: 0 = fp32 out (row-major), 1 = bf16 out (row-major),
//        2 = bf16 out transposed: out[col*M + row]  (used for V^T)
// ---------------------------------------------------------------------------
template <int OMODE>
__global__ __launch_bounds__(128) void gemm_bias_k(
    const __bf16* __restrict__ A, const __bf16* __restrict__ W,
    const float* __restrict__ bias, void* __restrict__ outp,
    int M, int N, int K)
{
  const int lane = threadIdx.x & 31;
  const int wave = threadIdx.x >> 5;
  const int l16  = lane & 15;
  const int hi   = lane >> 4;
  const int m0 = blockIdx.x * 128 + wave * 32;
  const int n0 = blockIdx.y * 64;

  int ar0 = m0 + l16;      if (ar0 > M - 1) ar0 = M - 1;   // clamp, guarded store
  int ar1 = m0 + 16 + l16; if (ar1 > M - 1) ar1 = M - 1;
  const __bf16* Arow0 = A + (size_t)ar0 * K;
  const __bf16* Arow1 = A + (size_t)ar1 * K;

  v8f zero = {0.f, 0.f, 0.f, 0.f, 0.f, 0.f, 0.f, 0.f};
  v8f acc[2][4];
#pragma unroll
  for (int r = 0; r < 2; ++r)
#pragma unroll
    for (int t = 0; t < 4; ++t) acc[r][t] = zero;

  for (int kk = 0; kk < K; kk += 32) {
    v16bf a0 = frag_bf(Arow0 + kk, hi);
    v16bf a1 = frag_bf(Arow1 + kk, hi);
#pragma unroll
    for (int t = 0; t < 4; ++t) {
      v16bf b = frag_bf(W + (size_t)(n0 + t * 16 + l16) * K + kk, hi);
      acc[0][t] = wmma_bf16(a0, b, acc[0][t]);
      acc[1][t] = wmma_bf16(a1, b, acc[1][t]);
    }
  }

#pragma unroll
  for (int r = 0; r < 2; ++r)
#pragma unroll
    for (int t = 0; t < 4; ++t)
#pragma unroll
      for (int v = 0; v < 8; ++v) {
        int row = m0 + r * 16 + v + hi * 8;   // C layout: lane<16 -> M=v, else v+8
        int col = n0 + t * 16 + l16;
        float val = acc[r][t][v] + bias[col];
        if (row < M) {
          if (OMODE == 0)      ((float*)outp)[(size_t)row * N + col] = val;
          else if (OMODE == 1) ((__bf16*)outp)[(size_t)row * N + col] = f2bf(val);
          else                 ((__bf16*)outp)[(size_t)col * M + row] = f2bf(val);
        }
      }
}

// ---------------------------------------------------------------------------
// Fused RMSNorm (over 1536, fp32 accumulation) + interleaved RoPE -> bf16
// One 256-thread block per token.  Input is bf16 (post-projection).
// ---------------------------------------------------------------------------
__global__ __launch_bounds__(256) void rmsnorm_rope_k(
    const __bf16* __restrict__ X, const float* __restrict__ G,
    const float* __restrict__ FC, const float* __restrict__ FS,
    __bf16* __restrict__ Out)
{
  __shared__ float red[256];
  const int p = blockIdx.x;
  const int tid = threadIdx.x;
  const __bf16* xr = X + (size_t)p * 1536;

  float ss = 0.f;
  for (int i = tid; i < 1536; i += 256) { float t = (float)xr[i]; ss += t * t; }
  red[tid] = ss;
  __syncthreads();
  for (int s = 128; s > 0; s >>= 1) {
    if (tid < s) red[tid] += red[tid + s];
    __syncthreads();
  }
  const float r = rsqrtf(red[0] * (1.0f / 1536.0f) + 1e-5f);

  const float* fcr = FC + (size_t)p * 128;    // cos at even indices
  const float* fsr = FS + (size_t)p * 128;    // sin at odd indices
  __bf16* orow = Out + (size_t)p * 1536;
  for (int pp = tid; pp < 768; pp += 256) {   // pair index over full DIM
    int j = pp & 63;                          // pair index within head
    float e = (float)xr[2 * pp]     * r * G[2 * pp];
    float o = (float)xr[2 * pp + 1] * r * G[2 * pp + 1];
    float c = fcr[2 * j];
    float s = fsr[2 * j + 1];
    orow[2 * pp]     = f2bf(e * c - o * s);
    orow[2 * pp + 1] = f2bf(e * s + o * c);
  }
}

// ---------------------------------------------------------------------------
// Block-sparse flash attention.
// grid = (q_tiles=25, TQ=2, HEADS=12), block = 128 (4 waves).
// Each wave: 16 query rows x full DH=128, online softmax over 2 selected
// key blocks, key tiles of 32 (QK^T: 8 wmma, PV: 8 wmma per tile).
// Q/K: bf16 [token][1536] row-major.  Vt: bf16 [head*128+dh][KTOT].
// ---------------------------------------------------------------------------
__global__ __launch_bounds__(128) void attn_k(
    const __bf16* __restrict__ Q, const __bf16* __restrict__ Kc,
    const __bf16* __restrict__ Vt, const int* __restrict__ sel,
    __bf16* __restrict__ Out)
{
  constexpr int HWc = 1560, KTOTc = 12480, DIMc = 1536;
  const int lane = threadIdx.x & 31;
  const int wave = threadIdx.x >> 5;
  const int l16  = lane & 15;
  const int hi   = lane >> 4;
  const int h  = blockIdx.z;
  const int tq = blockIdx.y;
  const int qt = blockIdx.x;

  __shared__ unsigned short Plds[4 * 512];    // 16x32 bf16 P tile per wave
  unsigned short* pb = Plds + wave * 512;

  int qr  = qt * 64 + wave * 16 + l16;
  int qrc = qr < HWc ? qr : HWc - 1;
  const __bf16* qrow = Q + (size_t)(tq * HWc + qrc) * DIMc + h * 128;
  v16bf qa[4];
#pragma unroll
  for (int c = 0; c < 4; ++c) qa[c] = frag_bf(qrow + c * 32, hi);

  v8f zero = {0.f, 0.f, 0.f, 0.f, 0.f, 0.f, 0.f, 0.f};
  v8f o[8];
#pragma unroll
  for (int nt = 0; nt < 8; ++nt) o[nt] = zero;
  float mrow[8], lrow[8];
#pragma unroll
  for (int v = 0; v < 8; ++v) { mrow[v] = -1e30f; lrow[v] = 0.f; }

  const float scale = 0.08838834764831845f;   // 1/sqrt(128)

  for (int kb = 0; kb < 2; ++kb) {
    int sb = sel[tq * 2 + kb];                // wave-uniform
    if (sb < 0) continue;                     // masked block: contributes nothing
    const size_t kbase = (size_t)sb * HWc;

    for (int kt = 0; kt < 49; ++kt) {         // 49 * 32 = 1568 >= 1560 keys
      // ---- scores: S[sub] = Q(16x128) @ K^T(128x16), sub = key half-tile
      v8f S[2];
#pragma unroll
      for (int sub = 0; sub < 2; ++sub) {
        int kl  = kt * 32 + sub * 16 + l16;
        int klc = kl < HWc ? kl : HWc - 1;
        const __bf16* krow = Kc + (kbase + (size_t)klc) * DIMc + h * 128;
        v8f s = zero;
#pragma unroll
        for (int c = 0; c < 4; ++c)
          s = wmma_bf16(qa[c], frag_bf(krow + c * 32, hi), s);
        float mul = (kl < HWc) ? scale : 0.f;
        float add = (kl < HWc) ? 0.f : -1e30f;
#pragma unroll
        for (int v = 0; v < 8; ++v) s[v] = s[v] * mul + add;
        S[sub] = s;
      }

      // ---- online softmax (row stats live per C-VGPR, reduced over 16 lanes)
      float corr[8];
#pragma unroll
      for (int v = 0; v < 8; ++v) {
        float t = fmaxf(S[0][v], S[1][v]);
#pragma unroll
        for (int m = 1; m < 16; m <<= 1) t = fmaxf(t, __shfl_xor(t, m, 32));
        float mn = fmaxf(mrow[v], t);
        float c0 = __expf(mrow[v] - mn);
        float p0 = __expf(S[0][v] - mn);
        float p1 = __expf(S[1][v] - mn);
        float rs = p0 + p1;
#pragma unroll
        for (int m = 1; m < 16; m <<= 1) rs += __shfl_xor(rs, m, 32);
        lrow[v] = lrow[v] * c0 + rs;
        mrow[v] = mn;
        corr[v] = c0;
        int prow = v + hi * 8;                // C layout -> P[row][key] in LDS
        pb[prow * 32 + l16]      = bfbits(p0);
        pb[prow * 32 + 16 + l16] = bfbits(p1);
      }
#pragma unroll
      for (int nt = 0; nt < 8; ++nt)
#pragma unroll
        for (int v = 0; v < 8; ++v) o[nt][v] *= corr[v];

      // per-wave LDS bounce: wait for DS stores before reading A-layout frag
      asm volatile("s_wait_dscnt 0" ::: "memory");

      BF16x16 pu;
      const __bf16* pbb = (const __bf16*)pb;
      pu.h[0] = *(const v8bf*)(pbb + l16 * 32 + hi * 8);
      pu.h[1] = *(const v8bf*)(pbb + l16 * 32 + 16 + hi * 8);
      v16bf pa = pu.v;

      // ---- PV: O(16x128) += P(16x32) @ V(32x128) via pre-transposed V
      const size_t kcol = kbase + (size_t)kt * 32;
#pragma unroll
      for (int nt = 0; nt < 8; ++nt) {
        const __bf16* vb = Vt + (size_t)(h * 128 + nt * 16 + l16) * KTOTc + kcol;
        BF16x16 bu;
        bu.h[0] = *(const v8bf*)(vb + hi * 8);
        bu.h[1] = *(const v8bf*)(vb + 16 + hi * 8);
        o[nt] = wmma_bf16(pa, bu.v, o[nt]);
      }
    }
  }

  float inv[8];
#pragma unroll
  for (int v = 0; v < 8; ++v) inv[v] = 1.f / lrow[v];  // block 0 always valid
#pragma unroll
  for (int nt = 0; nt < 8; ++nt)
#pragma unroll
    for (int v = 0; v < 8; ++v) {
      int row = qt * 64 + wave * 16 + v + hi * 8;
      if (row < HWc) {
        size_t tok = (size_t)tq * HWc + row;
        Out[tok * DIMc + h * 128 + nt * 16 + l16] = f2bf(o[nt][v] * inv[v]);
      }
    }
}

// ---------------------------------------------------------------------------
// Host launcher
// ---------------------------------------------------------------------------
extern "C" void kernel_launch(void* const* d_in, const int* in_sizes, int n_in,
                              void* d_out, int out_size, void* d_ws, size_t ws_size,
                              hipStream_t stream) {
  (void)in_sizes; (void)n_in; (void)out_size; (void)ws_size;
  const float* hidden  = (const float*)d_in[0];
  const float* history = (const float*)d_in[1];
  const float* fcq = (const float*)d_in[2];
  const float* fsq = (const float*)d_in[3];
  const float* fck = (const float*)d_in[4];
  const float* fsk = (const float*)d_in[5];
  const int*   sel = (const int*)d_in[6];
  const float* Wq = (const float*)d_in[7];
  const float* bq = (const float*)d_in[8];
  const float* Wk = (const float*)d_in[9];
  const float* bk = (const float*)d_in[10];
  const float* Wv = (const float*)d_in[11];
  const float* bv = (const float*)d_in[12];
  const float* Wo = (const float*)d_in[13];
  const float* bo = (const float*)d_in[14];
  const float* gq = (const float*)d_in[15];
  const float* gk = (const float*)d_in[16];
  float* out = (float*)d_out;

  const size_t QT = 3120, KT = 12480, D = 1536;
  char* ws = (char*)d_ws;
  size_t off = 0;
  auto alloc = [&](size_t bytes) { size_t r = off; off += (bytes + 255) & ~(size_t)255; return r; };
  size_t o_hidbf  = alloc(QT * D * 2);            // bf16 hidden
  size_t o_histbf = alloc(KT * D * 2);            // bf16 history
  size_t o_wbf    = alloc(D * D * 2);             // bf16 staged weight (reused 4x)
  size_t o_qpre   = alloc(QT * D * 2 + 1024);     // bf16 q_pre; later bf16 attn out
  size_t o_kpre   = alloc(KT * D * 2 + 1024);     // bf16 k_pre; later bf16 V^T (+OOB pad)
  size_t o_qbf    = alloc(QT * D * 2);
  size_t o_kbf    = alloc(KT * D * 2);

  __bf16* hidbf  = (__bf16*)(ws + o_hidbf);
  __bf16* histbf = (__bf16*)(ws + o_histbf);
  __bf16* wbf    = (__bf16*)(ws + o_wbf);
  __bf16* qpre   = (__bf16*)(ws + o_qpre);
  __bf16* kpre   = (__bf16*)(ws + o_kpre);
  __bf16* qbf    = (__bf16*)(ws + o_qbf);
  __bf16* kbf    = (__bf16*)(ws + o_kbf);
  __bf16* vtb    = (__bf16*)(ws + o_kpre);        // aliases k_pre (consumed first)
  __bf16* abf    = (__bf16*)(ws + o_qpre);        // aliases q_pre (consumed first)

  const unsigned gW   = (unsigned)(D * D / 2048);   // weight cvt grid
  const unsigned gHid = (unsigned)(QT * D / 2048);
  const unsigned gHis = (unsigned)(KT * D / 2048);

  // one-time activations -> bf16
  cvt_bf16_k<<<gHid, 256, 0, stream>>>(hidden,  hidbf,  (long)(QT * D));
  cvt_bf16_k<<<gHis, 256, 0, stream>>>(history, histbf, (long)(KT * D));

  // Q projection (bf16 out, RMSNorm consumes bf16)
  cvt_bf16_k<<<gW, 256, 0, stream>>>(Wq, wbf, (long)(D * D));
  gemm_bias_k<1><<<dim3(25, 24), 128, 0, stream>>>(hidbf, wbf, bq, qpre, (int)QT, (int)D, (int)D);
  // K projection
  cvt_bf16_k<<<gW, 256, 0, stream>>>(Wk, wbf, (long)(D * D));
  gemm_bias_k<1><<<dim3(98, 24), 128, 0, stream>>>(histbf, wbf, bk, kpre, (int)KT, (int)D, (int)D);
  // RMSNorm + RoPE -> bf16
  rmsnorm_rope_k<<<dim3((unsigned)QT), 256, 0, stream>>>(qpre, gq, fcq, fsq, qbf);
  rmsnorm_rope_k<<<dim3((unsigned)KT), 256, 0, stream>>>(kpre, gk, fck, fsk, kbf);
  // V projection, stored transposed bf16 [dh][key] (overwrites k_pre region)
  cvt_bf16_k<<<gW, 256, 0, stream>>>(Wv, wbf, (long)(D * D));
  gemm_bias_k<2><<<dim3(98, 24), 128, 0, stream>>>(histbf, wbf, bv, vtb, (int)KT, (int)D, (int)D);
  // Block-sparse flash attention -> bf16 (overwrites q_pre region)
  attn_k<<<dim3(25, 2, 12), 128, 0, stream>>>(qbf, kbf, vtb, sel, abf);
  // Output projection (fp32 result)
  cvt_bf16_k<<<gW, 256, 0, stream>>>(Wo, wbf, (long)(D * D));
  gemm_bias_k<0><<<dim3(25, 24), 128, 0, stream>>>(abf, Wo == nullptr ? wbf : wbf, bo, out, (int)QT, (int)D, (int)D);
}
